// GraphLearner_76922864271377
// MI455X (gfx1250) — matrix operand order repaired and checked by reading, not compiled, not run on hardware
//
#include <hip/hip_runtime.h>
#include <hip/hip_bf16.h>

// ---------------------------------------------------------------------------
// GraphLearner on MI455X (gfx1250):
//   S = (1/P) * W * W^T,  W[n, p*D+d] = normalize_d(features[n,:]*weight[p,:])
//   adj = where(S > 0.65, S, 0)   (S symmetric by construction)
//
// N=8192, D=256, P=8  =>  GEMM: M=N=8192, K=P*D=2048.
// W (f16) = 32 MB -> lives in the 192 MB L2; HBM traffic ~= 256 MB output.
// Compute-bound => v_wmma_f32_16x16x32_f16 with fp32 accumulation, and we
// exploit symmetry: only the upper block-triangle is computed (2x FLOP cut);
// off-diagonal tiles are stored twice (in place + transposed b128 stores).
// ---------------------------------------------------------------------------

typedef __attribute__((ext_vector_type(16))) _Float16 v16h;
typedef __attribute__((ext_vector_type(8)))  _Float16 v8h;
typedef __attribute__((ext_vector_type(8)))  float    v8f;

#define GL_N 8192
#define GL_D 256
#define GL_P 8
#define GL_K 2048              // P*D
#define GL_EPS 1e-12f
#define GL_THRESH 0.65f
#define GL_INV_P 0.125f

// ---------------------------------------------------------------------------
// Kernel 1: build W[n, p*D + d] = f[n,d]*w[p,d] / max(||f[n,:]*w[p,:]||, eps)
// One wave32 per (n,p) pair; each lane owns 8 consecutive d values.
// ---------------------------------------------------------------------------
__global__ __launch_bounds__(256) void gl_build_w(
    const float* __restrict__ feat,     // [N, D]
    const float* __restrict__ weight,   // [P, D]
    _Float16* __restrict__ W)           // [N, K]
{
  const int gwave = (blockIdx.x * blockDim.x + threadIdx.x) >> 5;  // 0..N*P-1
  const int lane  = threadIdx.x & 31;
  const int n = gwave >> 3;          // node
  const int p = gwave & (GL_P - 1);  // perspective

  const float4* fr = (const float4*)(feat   + (size_t)n * GL_D);
  const float4* wr = (const float4*)(weight + (size_t)p * GL_D);
  // lane covers d = lane*8 .. lane*8+7
  const float4 f0 = fr[lane * 2 + 0];
  const float4 f1 = fr[lane * 2 + 1];
  const float4 w0 = wr[lane * 2 + 0];
  const float4 w1 = wr[lane * 2 + 1];

  float c[8];
  c[0] = f0.x * w0.x; c[1] = f0.y * w0.y; c[2] = f0.z * w0.z; c[3] = f0.w * w0.w;
  c[4] = f1.x * w1.x; c[5] = f1.y * w1.y; c[6] = f1.z * w1.z; c[7] = f1.w * w1.w;

  float ss = 0.0f;
#pragma unroll
  for (int i = 0; i < 8; ++i) ss += c[i] * c[i];
  // wave32 butterfly reduction
#pragma unroll
  for (int m = 16; m >= 1; m >>= 1) ss += __shfl_xor(ss, m, 32);

  const float scale = 1.0f / fmaxf(sqrtf(ss), GL_EPS);

  v8h outv;
#pragma unroll
  for (int i = 0; i < 8; ++i) outv[i] = (_Float16)(c[i] * scale);

  *(v8h*)(W + (size_t)n * GL_K + p * GL_D + lane * 8) = outv;
}

// ---------------------------------------------------------------------------
// Kernel 2: S = (1/P) * W * W^T with threshold epilogue, upper triangle only.
// Block = 256 threads = 8 waves, computes a 128(M) x 128(N) tile where
// blockRow <= blockCol; off-diagonal tiles are also stored transposed.
// Wave grid 4(M) x 2(N): each wave owns 32(M) x 64(N) = 2x4 WMMA tiles.
//
// Fragment loads use contiguous k order (same for A and B, not the ISA table order):
//   A lane(m=lane&15): halves = K[k0..k0+7], K[k0+8..k0+15], k0=(lane>>4)*16
//       -> two contiguous 16B loads from row-major W.
//   B lane(n=lane&15): halves = K[(lane>>4)*16 + 0..15]
//       -> B = W^T, so column n of B is row n of W: one contiguous 32B load.
// ---------------------------------------------------------------------------
__global__ __launch_bounds__(256) void gl_gemm_thresh(
    const _Float16* __restrict__ W,   // [N, K]
    float* __restrict__ out)          // [N, N]
{
  const int bRow = blockIdx.y;              // M block
  const int bCol = blockIdx.x;              // N block
  if (bRow > bCol) return;                  // symmetric: skip lower triangle
  const bool offDiag = (bRow != bCol);

  const int lane   = threadIdx.x & 31;
  const int waveId = threadIdx.x >> 5;      // 0..7
  const int laneLo = lane & 15;
  const int laneHi = lane >> 4;             // 0 or 1

  const int waveM = waveId & 3;             // 4 waves down M
  const int waveN = waveId >> 2;            // 2 waves across N
  const int wm = bRow * 128 + waveM * 32;   // wave M base (32 rows)
  const int wn = bCol * 128 + waveN * 64;   // wave N base (64 cols)

  // A-side row pointers (2 M-subtiles)
  const _Float16* arow0 = W + (size_t)(wm +  0 + laneLo) * GL_K;
  const _Float16* arow1 = W + (size_t)(wm + 16 + laneLo) * GL_K;
  // B-side row pointers (4 N-subtiles); B column n == W row n
  const _Float16* brow0 = W + (size_t)(wn +  0 + laneLo) * GL_K;
  const _Float16* brow1 = W + (size_t)(wn + 16 + laneLo) * GL_K;
  const _Float16* brow2 = W + (size_t)(wn + 32 + laneLo) * GL_K;
  const _Float16* brow3 = W + (size_t)(wn + 48 + laneLo) * GL_K;

  v8f acc[2][4] = {};   // 64 f32 accumulator VGPRs

  for (int kb = 0; kb < GL_K; kb += 32) {
    const int ak = kb + laneHi * 16;   // A half-chunk base
    const int bk = kb + laneHi * 16;   // B chunk base

    // A fragments: two 16B loads each, fused into v16h
    v8h a0lo = *(const v8h*)(arow0 + ak);
    v8h a0hi = *(const v8h*)(arow0 + ak + 8);
    v8h a1lo = *(const v8h*)(arow1 + ak);
    v8h a1hi = *(const v8h*)(arow1 + ak + 8);
    v16h a0 = __builtin_shufflevector(a0lo, a0hi, 0,1,2,3,4,5,6,7,8,9,10,11,12,13,14,15);
    v16h a1 = __builtin_shufflevector(a1lo, a1hi, 0,1,2,3,4,5,6,7,8,9,10,11,12,13,14,15);

    // B fragments: one contiguous 32B load each
    v16h b0 = *(const v16h*)(brow0 + bk);
    v16h b1 = *(const v16h*)(brow1 + bk);
    v16h b2 = *(const v16h*)(brow2 + bk);
    v16h b3 = *(const v16h*)(brow3 + bk);

    // 8 x v_wmma_f32_16x16x32_f16
    acc[0][0] = __builtin_amdgcn_wmma_f32_16x16x32_f16(false, a0, false, b0, (short)0, acc[0][0], false, false);
    acc[0][1] = __builtin_amdgcn_wmma_f32_16x16x32_f16(false, a0, false, b1, (short)0, acc[0][1], false, false);
    acc[0][2] = __builtin_amdgcn_wmma_f32_16x16x32_f16(false, a0, false, b2, (short)0, acc[0][2], false, false);
    acc[0][3] = __builtin_amdgcn_wmma_f32_16x16x32_f16(false, a0, false, b3, (short)0, acc[0][3], false, false);
    acc[1][0] = __builtin_amdgcn_wmma_f32_16x16x32_f16(false, a1, false, b0, (short)0, acc[1][0], false, false);
    acc[1][1] = __builtin_amdgcn_wmma_f32_16x16x32_f16(false, a1, false, b1, (short)0, acc[1][1], false, false);
    acc[1][2] = __builtin_amdgcn_wmma_f32_16x16x32_f16(false, a1, false, b2, (short)0, acc[1][2], false, false);
    acc[1][3] = __builtin_amdgcn_wmma_f32_16x16x32_f16(false, a1, false, b3, (short)0, acc[1][3], false, false);
  }

  // Epilogue: scale by 1/P, threshold at 0.65, store fp32.
  // C/D layout: VGPR v, lanes 0-15: (M=v, N=lane); lanes 16-31: (M=v+8, N=lane-16).
  // Off-diagonal blocks also store the transposed tile: per lane the 8 acc
  // values are consecutive output columns -> two aligned float4 stores.
#pragma unroll
  for (int mi = 0; mi < 2; ++mi) {
#pragma unroll
    for (int ni = 0; ni < 4; ++ni) {
      const int r0 = wm + mi * 16 + laneHi * 8;
      const int c  = wn + ni * 16 + laneLo;

      float vals[8];
#pragma unroll
      for (int v = 0; v < 8; ++v) {
        float s = acc[mi][ni][v] * GL_INV_P;
        vals[v] = (s > GL_THRESH) ? s : 0.0f;
      }

      // normal orientation: one float per (row, c), rows r0..r0+7
#pragma unroll
      for (int v = 0; v < 8; ++v)
        out[(size_t)(r0 + v) * GL_N + c] = vals[v];

      // mirrored tile: row c, columns r0..r0+7 (16B-aligned vector stores)
      if (offDiag) {
        float4 lo, hi;
        lo.x = vals[0]; lo.y = vals[1]; lo.z = vals[2]; lo.w = vals[3];
        hi.x = vals[4]; hi.y = vals[5]; hi.z = vals[6]; hi.w = vals[7];
        float* mrow = out + (size_t)c * GL_N + r0;
        *(float4*)(mrow)     = lo;
        *(float4*)(mrow + 4) = hi;
      }
    }
  }
}

// ---------------------------------------------------------------------------
extern "C" void kernel_launch(void* const* d_in, const int* in_sizes, int n_in,
                              void* d_out, int out_size, void* d_ws, size_t ws_size,
                              hipStream_t stream) {
  const float* features = (const float*)d_in[0];   // [8192, 256] fp32
  const float* weight   = (const float*)d_in[1];   // [8, 256]    fp32
  float*       out      = (float*)d_out;           // [8192, 8192] fp32
  _Float16*    W        = (_Float16*)d_ws;         // [8192, 2048] f16 = 32 MB

  // Phase 1: N*P = 65536 waves, 8 waves per 256-thread block -> 8192 blocks
  gl_build_w<<<dim3(8192), dim3(256), 0, stream>>>(features, weight, W);

  // Phase 2: 128x128 tiles, upper block-triangle computed, mirror-stored.
  // Launch full 64x64 grid; lower-triangle blocks exit immediately.
  gl_gemm_thresh<<<dim3(64, 64), dim3(256), 0, stream>>>(W, out);
}
